// SparseMPNNLayer_31808527794619
// MI455X (gfx1250) — matrix operand order repaired and checked by reading, not compiled, not run on hardware
//
#include <hip/hip_runtime.h>

// ---------------------------------------------------------------------------
// CDNA5 (gfx1250, wave32) WMMA bf16 implementation of a bipartite MPNN layer.
// Memory-bound problem (~1.1 GB traffic @ 23.3 TB/s vs ~70 GFLOP), so all
// GEMMs use v_wmma_f32_16x16x32_bf16 (f32 accumulate), gathers are whole
// 256B rows, weights are pre-transposed to bf16 [N][K] so each B-fragment is
// two 16-byte loads, and f32->bf16 activation conversion packs two
// values with a single v_perm_b32 per pair.
// ---------------------------------------------------------------------------

typedef __attribute__((ext_vector_type(16))) __bf16    v16bf;
typedef __attribute__((ext_vector_type(8)))  __bf16    v8bf;
typedef __attribute__((ext_vector_type(8)))  float     v8f;
typedef __attribute__((ext_vector_type(4)))  float     v4f;
typedef __attribute__((ext_vector_type(8)))  unsigned  v8u;

// RNE f32->bf16 (used only in the one-time weight conversion kernel).
__device__ __forceinline__ __bf16 f2bf_rne(float f) {
  unsigned u = __builtin_bit_cast(unsigned, f);
  unsigned r = u + 0x7FFFu + ((u >> 16) & 1u);
  unsigned short h = (unsigned short)(r >> 16);
  return __builtin_bit_cast(__bf16, h);
}

// Fast packed truncating f32x2 -> bf16x2: one v_perm_b32 selecting the two
// high halves ({hi[3],hi[2],lo[3],lo[2]} byte selectors = 0x07060302).
__device__ __forceinline__ unsigned pk2bf(float lo, float hi) {
  unsigned a = __builtin_bit_cast(unsigned, lo);
  unsigned b = __builtin_bit_cast(unsigned, hi);
#if __has_builtin(__builtin_amdgcn_perm)
  return __builtin_amdgcn_perm(b, a, 0x07060302u);
#else
  return (b & 0xFFFF0000u) | (a >> 16);
#endif
}

// Truncating f32 -> raw bf16 bits (folds into d16_hi-style stores).
__device__ __forceinline__ unsigned short bf_bits(float f) {
  return (unsigned short)(__builtin_bit_cast(unsigned, f) >> 16);
}

__device__ __forceinline__ v8f wmma_bf16(v16bf a, v16bf b, v8f c) {
  // (neg_a, A, neg_b, B, c_mod, C, reuse_a, reuse_b)
  return __builtin_amdgcn_wmma_f32_16x16x32_bf16(false, a, false, b,
                                                 (short)0, c, false, false);
}

// A-fragment (16x32 bf16) built from an f32 feature row.
// Lane L holds row M = L%16; element i holds K = (i/8)*16 + (L/16)*8 + (i%8)
// => two contiguous 8-float runs at base+lh*8 and base+16+lh*8.
__device__ __forceinline__ v16bf load_a_f32(const float* __restrict__ base,
                                            int lh) {
  const v4f* q0 = (const v4f*)(base + lh * 8);
  const v4f* q1 = (const v4f*)(base + 16 + lh * 8);
  v4f x0 = q0[0], x1 = q0[1], x2 = q1[0], x3 = q1[1];
  v8u pk;
  pk[0] = pk2bf(x0[0], x0[1]);
  pk[1] = pk2bf(x0[2], x0[3]);
  pk[2] = pk2bf(x1[0], x1[1]);
  pk[3] = pk2bf(x1[2], x1[3]);
  pk[4] = pk2bf(x2[0], x2[1]);
  pk[5] = pk2bf(x2[2], x2[3]);
  pk[6] = pk2bf(x3[0], x3[1]);
  pk[7] = pk2bf(x3[2], x3[3]);
  return __builtin_bit_cast(v16bf, pk);
}

__device__ __forceinline__ v16bf load_a_f32_scaled(const float* __restrict__ base,
                                                   int lh, float s) {
  const v4f* q0 = (const v4f*)(base + lh * 8);
  const v4f* q1 = (const v4f*)(base + 16 + lh * 8);
  v4f x0 = q0[0] * s, x1 = q0[1] * s, x2 = q1[0] * s, x3 = q1[1] * s;
  v8u pk;
  pk[0] = pk2bf(x0[0], x0[1]);
  pk[1] = pk2bf(x0[2], x0[3]);
  pk[2] = pk2bf(x1[0], x1[1]);
  pk[3] = pk2bf(x1[2], x1[3]);
  pk[4] = pk2bf(x2[0], x2[1]);
  pk[5] = pk2bf(x2[2], x2[3]);
  pk[6] = pk2bf(x3[0], x3[1]);
  pk[7] = pk2bf(x3[2], x3[3]);
  return __builtin_bit_cast(v16bf, pk);
}

// A-fragment from bf16 bits stored in LDS (layer-2 input after ReLU).
__device__ __forceinline__ v16bf load_a_lds(const unsigned short* row, int k0,
                                            int lh) {
  v8bf x0 = *(const v8bf*)(row + k0 + lh * 8);
  v8bf x1 = *(const v8bf*)(row + k0 + 16 + lh * 8);
  return __builtin_shufflevector(x0, x1, 0, 1, 2, 3, 4, 5, 6, 7,
                                 8, 9, 10, 11, 12, 13, 14, 15);
}

// B-fragment (32x16 bf16) from pre-transposed weights Wt[N][K].
// Lane L holds col N = n; element i holds K = k0 + (i/8)*16 + (L/16)*8 + (i%8)
// => two contiguous 8-bf16 runs = two 16-byte loads.
__device__ __forceinline__ v16bf load_b(const __bf16* __restrict__ Wt, int ldk,
                                        int n, int k0, int lh) {
  const __bf16* p = Wt + (size_t)n * ldk + k0 + lh * 8;
  v8bf x0 = *(const v8bf*)p;
  v8bf x1 = *(const v8bf*)(p + 16);
  return __builtin_shufflevector(x0, x1, 0, 1, 2, 3, 4, 5, 6, 7,
                                 8, 9, 10, 11, 12, 13, 14, 15);
}

// ---------------------------------------------------------------------------
// Edge message kernel: msg = MLP([featA[idxA], featB[idxB], efeat]) ;
// atomic scatter-add into accum[sIdx].   K=192 -> 128 (ReLU) -> 64.
// One wave per 16-edge tile; 48 + 16 WMMAs per tile.
// ---------------------------------------------------------------------------
__global__ void __launch_bounds__(256)
edge_msg_kernel(const float* __restrict__ featA, const long long* __restrict__ idxA,
                const float* __restrict__ featB, const long long* __restrict__ idxB,
                const float* __restrict__ efeat,
                const long long* __restrict__ sIdx,
                const __bf16* __restrict__ W1t, const float* __restrict__ b1,
                const __bf16* __restrict__ W2t, const float* __restrict__ b2,
                float* __restrict__ accum, int E) {
  __shared__ __align__(16) unsigned short sh[8][16][128];  // per-wave hidden (32KB)

  const int lane = threadIdx.x & 31;
  const int wib  = threadIdx.x >> 5;
  const int wave = blockIdx.x * 8 + wib;
  const int m0   = wave * 16;
  const bool active = (m0 + 16) <= E;
  const int lh  = lane >> 4;   // which 16-lane half
  const int c16 = lane & 15;

  __builtin_prefetch(W1t, 0, 0);
  __builtin_prefetch(W2t, 0, 0);

  v8f acc[8] = {};
  if (active) {
    const int rowE = m0 + c16;                 // edge this lane gathers for A
    const long long iA = idxA[rowE];
    const long long iB = idxB[rowE];
    const float* pA = featA + iA * 64;
    const float* pB = featB + iB * 64;
    const float* pE = efeat + (long long)rowE * 64;
#pragma unroll
    for (int kt = 0; kt < 6; ++kt) {
      const int k0 = kt * 32;
      const float* base = (k0 < 64)  ? (pA + k0)
                        : (k0 < 128) ? (pB + (k0 - 64))
                                     : (pE + (k0 - 128));
      v16bf a = load_a_f32(base, lh);
#pragma unroll
      for (int nt = 0; nt < 8; ++nt) {
        v16bf b = load_b(W1t, 192, nt * 16 + c16, k0, lh);
        acc[nt] = wmma_bf16(a, b, acc[nt]);
      }
    }
    // bias + ReLU, stash hidden (16x128) in LDS as bf16 bits
#pragma unroll
    for (int nt = 0; nt < 8; ++nt) {
      const int n = nt * 16 + c16;
      const float bias = b1[n];
#pragma unroll
      for (int r = 0; r < 8; ++r) {
        float v = acc[nt][r] + bias;
        v = v > 0.f ? v : 0.f;
        sh[wib][lh * 8 + r][n] = bf_bits(v);
      }
    }
  }
  __syncthreads();
  if (active) {
    v8f acc2[4] = {};
#pragma unroll
    for (int kt = 0; kt < 4; ++kt) {
      const int k0 = kt * 32;
      v16bf a = load_a_lds(&sh[wib][c16][0], k0, lh);
#pragma unroll
      for (int nt = 0; nt < 4; ++nt) {
        v16bf b = load_b(W2t, 128, nt * 16 + c16, k0, lh);
        acc2[nt] = wmma_bf16(a, b, acc2[nt]);
      }
    }
    long long sc[8];
#pragma unroll
    for (int r = 0; r < 8; ++r) sc[r] = sIdx[m0 + lh * 8 + r];
#pragma unroll
    for (int nt = 0; nt < 4; ++nt) {
      const int n = nt * 16 + c16;
      const float bias = b2[n];
#pragma unroll
      for (int r = 0; r < 8; ++r)
        atomicAdd(&accum[sc[r] * 64 + n], acc2[nt][r] + bias);
    }
  }
}

// ---------------------------------------------------------------------------
// Node update kernel: out = MLP([nodeFeat, msg*scale]) ; K=128 -> 64 -> 64.
// scale = 1/S (degp==null) or 1/max(deg,1) per row (degp!=null).
// ---------------------------------------------------------------------------
__global__ void __launch_bounds__(256)
node_update_kernel(const float* __restrict__ nodeFeat,
                   const float* __restrict__ msg,
                   const float* __restrict__ degp,   // nullable
                   const int* __restrict__ Sp,       // nullable
                   const __bf16* __restrict__ W1t, const float* __restrict__ b1,
                   const __bf16* __restrict__ W2t, const float* __restrict__ b2,
                   float* __restrict__ out, int N) {
  __shared__ __align__(16) unsigned short sh[8][16][64];  // per-wave hidden (16KB)

  const int lane = threadIdx.x & 31;
  const int wib  = threadIdx.x >> 5;
  const int wave = blockIdx.x * 8 + wib;
  const int m0   = wave * 16;
  const bool active = (m0 + 16) <= N;
  const int lh  = lane >> 4;
  const int c16 = lane & 15;

  v8f acc[4] = {};
  if (active) {
    const int row = m0 + c16;
    float scale;
    if (degp) {
      float d = degp[row];
      scale = 1.0f / (d > 1.0f ? d : 1.0f);
    } else {
      scale = 1.0f / (float)(*Sp);
    }
    const float* pn = nodeFeat + (long long)row * 64;
    const float* pm = msg + (long long)row * 64;
#pragma unroll
    for (int kt = 0; kt < 4; ++kt) {
      const int k0 = kt * 32;
      v16bf a = (k0 < 64) ? load_a_f32(pn + k0, lh)
                          : load_a_f32_scaled(pm + (k0 - 64), lh, scale);
#pragma unroll
      for (int nt = 0; nt < 4; ++nt) {
        v16bf b = load_b(W1t, 128, nt * 16 + c16, k0, lh);
        acc[nt] = wmma_bf16(a, b, acc[nt]);
      }
    }
#pragma unroll
    for (int nt = 0; nt < 4; ++nt) {
      const int n = nt * 16 + c16;
      const float bias = b1[n];
#pragma unroll
      for (int r = 0; r < 8; ++r) {
        float v = acc[nt][r] + bias;
        v = v > 0.f ? v : 0.f;
        sh[wib][lh * 8 + r][n] = bf_bits(v);
      }
    }
  }
  __syncthreads();
  if (active) {
    v8f acc2[4] = {};
#pragma unroll
    for (int kt = 0; kt < 2; ++kt) {
      const int k0 = kt * 32;
      v16bf a = load_a_lds(&sh[wib][c16][0], k0, lh);
#pragma unroll
      for (int nt = 0; nt < 4; ++nt) {
        v16bf b = load_b(W2t, 64, nt * 16 + c16, k0, lh);
        acc2[nt] = wmma_bf16(a, b, acc2[nt]);
      }
    }
#pragma unroll
    for (int nt = 0; nt < 4; ++nt) {
      const int n = nt * 16 + c16;
      const float bias = b2[n];
#pragma unroll
      for (int r = 0; r < 8; ++r)
        out[(size_t)(m0 + lh * 8 + r) * 64 + n] = acc2[nt][r] + bias;
    }
  }
}

// f32 [K][N] row-major -> bf16 [N][K] (so B-fragments are contiguous).
__global__ void convert_w_kernel(const float* __restrict__ W,
                                 __bf16* __restrict__ Wt, int K, int N) {
  int idx = blockIdx.x * blockDim.x + threadIdx.x;
  if (idx < K * N) {
    int k = idx / N, n = idx % N;
    Wt[(size_t)n * K + k] = f2bf_rne(W[idx]);
  }
}

__global__ void deg_kernel(const long long* __restrict__ src,
                           float* __restrict__ deg, int E) {
  int e = blockIdx.x * blockDim.x + threadIdx.x;
  if (e < E) atomicAdd(&deg[src[e]], 1.0f);
}

// ---------------------------------------------------------------------------
extern "C" void kernel_launch(void* const* d_in, const int* in_sizes, int n_in,
                              void* d_out, int out_size, void* d_ws, size_t ws_size,
                              hipStream_t stream) {
  (void)n_in; (void)out_size; (void)ws_size;
  const int H = 64, NV = 100000, NU = 100000;
  const float* h_v    = (const float*)d_in[0];
  const float* h_u    = (const float*)d_in[1];
  const float* e_feat = (const float*)d_in[2];
  const long long* eidx = (const long long*)d_in[3];
  const int E = in_sizes[2] / H;
  const long long* src = eidx;       // edge_index[0]
  const long long* dst = eidx + E;   // edge_index[1]
  const int* Sp = (const int*)d_in[4];
  const float* a2u_w1 = (const float*)d_in[5];
  const float* a2u_b1 = (const float*)d_in[6];
  const float* a2u_w2 = (const float*)d_in[7];
  const float* a2u_b2 = (const float*)d_in[8];
  const float* u_w1   = (const float*)d_in[9];
  const float* u_b1   = (const float*)d_in[10];
  const float* u_w2   = (const float*)d_in[11];
  const float* u_b2   = (const float*)d_in[12];
  const float* u2a_w1 = (const float*)d_in[13];
  const float* u2a_b1 = (const float*)d_in[14];
  const float* u2a_w2 = (const float*)d_in[15];
  const float* u2a_b2 = (const float*)d_in[16];
  const float* a_w1   = (const float*)d_in[17];
  const float* a_b1   = (const float*)d_in[18];
  const float* a_w2   = (const float*)d_in[19];
  const float* a_b2   = (const float*)d_in[20];

  // ---- workspace layout (256B aligned slices) ----
  auto au = [](size_t x) { return (x + 255) & ~(size_t)255; };
  char* p = (char*)d_ws;
  float* m_u  = (float*)p; p += au((size_t)NU * 64 * 4);
  float* m_v  = (float*)p; p += au((size_t)NV * 64 * 4);
  float* degv = (float*)p; p += au((size_t)NV * 4);
  size_t zero_bytes = (size_t)(p - (char*)d_ws);
  __bf16* w_a2u1 = (__bf16*)p; p += au((size_t)192 * 128 * 2);
  __bf16* w_a2u2 = (__bf16*)p; p += au((size_t)128 * 64 * 2);
  __bf16* w_u1   = (__bf16*)p; p += au((size_t)128 * 64 * 2);
  __bf16* w_u2   = (__bf16*)p; p += au((size_t)64 * 64 * 2);
  __bf16* w_u2a1 = (__bf16*)p; p += au((size_t)192 * 128 * 2);
  __bf16* w_u2a2 = (__bf16*)p; p += au((size_t)128 * 64 * 2);
  __bf16* w_a1   = (__bf16*)p; p += au((size_t)128 * 64 * 2);
  __bf16* w_a2   = (__bf16*)p; p += au((size_t)64 * 64 * 2);

  hipMemsetAsync(d_ws, 0, zero_bytes, stream);

  // ---- weight conversion / transpose (f32 [K][N] -> bf16 [N][K]) ----
  auto cvblocks = [](int kn) { return (kn + 255) / 256; };
  convert_w_kernel<<<cvblocks(192 * 128), 256, 0, stream>>>(a2u_w1, w_a2u1, 192, 128);
  convert_w_kernel<<<cvblocks(128 * 64),  256, 0, stream>>>(a2u_w2, w_a2u2, 128, 64);
  convert_w_kernel<<<cvblocks(128 * 64),  256, 0, stream>>>(u_w1,   w_u1,   128, 64);
  convert_w_kernel<<<cvblocks(64 * 64),   256, 0, stream>>>(u_w2,   w_u2,   64,  64);
  convert_w_kernel<<<cvblocks(192 * 128), 256, 0, stream>>>(u2a_w1, w_u2a1, 192, 128);
  convert_w_kernel<<<cvblocks(128 * 64),  256, 0, stream>>>(u2a_w2, w_u2a2, 128, 64);
  convert_w_kernel<<<cvblocks(128 * 64),  256, 0, stream>>>(a_w1,   w_a1,   128, 64);
  convert_w_kernel<<<cvblocks(64 * 64),   256, 0, stream>>>(a_w2,   w_a2,   64,  64);

  deg_kernel<<<(E + 255) / 256, 256, 0, stream>>>(src, degv, E);

  const int etiles  = (E + 15) / 16;
  const int eblocks = (etiles + 7) / 8;
  const int utiles  = (NU + 15) / 16;
  const int ublocks = (utiles + 7) / 8;
  const int vtiles  = (NV + 15) / 16;
  const int vblocks = (vtiles + 7) / 8;

  float* h_v_out = (float*)d_out;
  float* h_u_out = (float*)d_out + (size_t)NV * 64;

  // a -> u messages: MLP([h_v[src], h_u[dst], e_feat]) scattered by dst into m_u
  edge_msg_kernel<<<eblocks, 256, 0, stream>>>(
      h_v, src, h_u, dst, e_feat, dst,
      w_a2u1, a2u_b1, w_a2u2, a2u_b2, m_u, E);

  // u node update: MLP([h_u, m_u/S]) -> h_u_out (written into d_out slot)
  node_update_kernel<<<ublocks, 256, 0, stream>>>(
      h_u, m_u, nullptr, Sp,
      w_u1, u_b1, w_u2, u_b2, h_u_out, NU);

  // u -> a messages: MLP([h_u_out[dst], h_v[src], e_feat]) scattered by src into m_v
  edge_msg_kernel<<<eblocks, 256, 0, stream>>>(
      h_u_out, dst, h_v, src, e_feat, src,
      w_u2a1, u2a_b1, w_u2a2, u2a_b2, m_v, E);

  // v node update: MLP([h_v, m_v/max(deg,1)]) -> h_v_out
  node_update_kernel<<<vblocks, 256, 0, stream>>>(
      h_v, m_v, degv, nullptr,
      w_a1, a_b1, w_a2, a_b2, h_v_out, NV);
}
